// HierarchicalPatternLayer_64287070487053
// MI455X (gfx1250) — matrix operand order repaired and checked by reading, not compiled, not run on hardware
//
#include <hip/hip_runtime.h>
#include <hip/hip_bf16.h>
#include <math.h>

typedef __attribute__((ext_vector_type(8)))  __bf16 v8bf;
typedef __attribute__((ext_vector_type(16))) __bf16 v16bf;
typedef __attribute__((ext_vector_type(8)))  float  v8f;
typedef __attribute__((ext_vector_type(8)))  unsigned short ushort8;
typedef __attribute__((ext_vector_type(4)))  int    int4v;

#define LDSP 40   // padded LDS row stride (32 data bf16 + 8 pad) = 80B

#define AS1 __attribute__((address_space(1)))
#define AS3 __attribute__((address_space(3)))

#if __has_builtin(__builtin_amdgcn_global_load_async_to_lds_b128)
#define HAVE_ASYNC_LDS 1
#else
#define HAVE_ASYNC_LDS 0
#endif

__device__ __forceinline__ unsigned short f2bf_rne(float f) {
    unsigned int u = __float_as_uint(f);
    unsigned int r = u + 0x7FFFu + ((u >> 16) & 1u);
    return (unsigned short)(r >> 16);
}

__device__ __forceinline__ void wait_async0() {
#if HAVE_ASYNC_LDS
#if __has_builtin(__builtin_amdgcn_s_wait_asynccnt)
    __builtin_amdgcn_s_wait_asynccnt(0);
#else
    asm volatile("s_wait_asynccnt 0" ::: "memory");
#endif
#endif
}

union Frag { v16bf v; v8bf h[2]; };

// FLAGS: 1=relu, 2=scale rows by scale[m], 4=store bf16 to Cbf, 8=store f32 to C
template<int FLAGS>
__global__ __launch_bounds__(256)
void gemm_bf16_wmma(const unsigned short* __restrict__ A, int lda,
                    const unsigned short* __restrict__ B, int ldb,
                    const float* __restrict__ bias,
                    float* __restrict__ C, unsigned short* __restrict__ Cbf, int ldc,
                    int K,
                    const float* __restrict__ scale)
{
    __shared__ unsigned short lsA[128 * LDSP];
    __shared__ unsigned short lsB[128 * LDSP];

    const int tid  = threadIdx.x;
    const int wave = tid >> 5;
    const int lane = tid & 31;
    const int half = lane >> 4;
    const int l16  = lane & 15;
    const int mBase = (wave >> 1) * 32;   // 4 wave-rows of 32
    const int nBase = (wave & 1) * 64;    // 2 wave-cols of 64

    const int blockN = blockIdx.x * 128;
    const int blockM = blockIdx.y * 128;

    v8f zero = {};
    v8f acc[2][4];
    for (int i = 0; i < 2; ++i)
        for (int j = 0; j < 4; ++j) acc[i][j] = zero;

    for (int kb = 0; kb < K; kb += 32) {
        // ---- stage A tile: 128x32 bf16, row-major, 512 chunks of 16B ----
        #pragma unroll
        for (int i = 0; i < 2; ++i) {
            int idx = tid + i * 256;
            int r   = idx >> 2;            // row 0..127
            int c   = idx & 3;             // 16B chunk 0..3
            const unsigned short* gp = A + (size_t)(blockM + r) * lda + kb + c * 8;
#if HAVE_ASYNC_LDS
            __builtin_amdgcn_global_load_async_to_lds_b128(
                (AS1 int4v*)(void*)gp,
                (AS3 int4v*)(void*)&lsA[r * LDSP + c * 8], 0, 0);
#else
            *(ushort8*)&lsA[r * LDSP + c * 8] = *(const ushort8*)gp;
#endif
        }
        // ---- stage B tile: 32x128 bf16, transposed -> lsB[n][k] ----
        #pragma unroll
        for (int i = 0; i < 2; ++i) {
            int idx = tid + i * 256;
            int r   = idx >> 4;            // k row 0..31
            int c8  = idx & 15;            // 8-elem chunk 0..15
            ushort8 v = *(const ushort8*)(B + (size_t)(kb + r) * ldb + blockN + c8 * 8);
            int n0 = c8 * 8;
            #pragma unroll
            for (int e = 0; e < 8; ++e) lsB[(n0 + e) * LDSP + r] = v[e];
        }
        wait_async0();
        __syncthreads();

        // A frag: lanes 0-15 hold K {0..7,16..23}, lanes 16-31 {8..15,24..31}
        Frag aF[2], bF[4];
        #pragma unroll
        for (int i = 0; i < 2; ++i) {
            int row = mBase + i * 16 + l16;
            int ko  = half * 8;
            aF[i].h[0] = *(const v8bf*)&lsA[row * LDSP + ko];
            aF[i].h[1] = *(const v8bf*)&lsA[row * LDSP + 16 + ko];
        }
        // B frag: lanes 0-15 hold K {0..7,16..23}, lanes 16-31 {8..15,24..31}
        #pragma unroll
        for (int j = 0; j < 4; ++j) {
            int col = nBase + j * 16 + l16;
            int ko  = half * 8;
            bF[j].h[0] = *(const v8bf*)&lsB[col * LDSP + ko];
            bF[j].h[1] = *(const v8bf*)&lsB[col * LDSP + 16 + ko];
        }
        #pragma unroll
        for (int i = 0; i < 2; ++i)
            #pragma unroll
            for (int j = 0; j < 4; ++j)
                acc[i][j] = __builtin_amdgcn_wmma_f32_16x16x32_bf16(
                    false, aF[i].v, false, bF[j].v, (short)0, acc[i][j], false, false);
        __syncthreads();
    }

    // epilogue: D layout = lanes 0-15: M=v, lanes 16-31: M=v+8; N=lane&15
    #pragma unroll
    for (int i = 0; i < 2; ++i) {
        #pragma unroll
        for (int j = 0; j < 4; ++j) {
            int nG = blockN + nBase + j * 16 + l16;
            float bv = bias[nG];
            #pragma unroll
            for (int v = 0; v < 8; ++v) {
                int mG = blockM + mBase + i * 16 + v + half * 8;
                float val = acc[i][j][v] + bv;
                if constexpr ((FLAGS & 1) != 0) val = fmaxf(val, 0.0f);
                if constexpr ((FLAGS & 2) != 0) val *= scale[mG];
                if constexpr ((FLAGS & 8) != 0) C[(size_t)mG * ldc + nG] = val;
                if constexpr ((FLAGS & 4) != 0) Cbf[(size_t)mG * ldc + nG] = f2bf_rne(val);
            }
        }
    }
}

__global__ __launch_bounds__(256)
void f32_to_bf16(const float* __restrict__ src, unsigned short* __restrict__ dst, int n8)
{
    int i = blockIdx.x * 256 + threadIdx.x;
    if (i >= n8) return;
    const float4* s = (const float4*)src + (size_t)i * 2;
    float4 a = s[0], b = s[1];
    ushort8 o;
    o[0] = f2bf_rne(a.x); o[1] = f2bf_rne(a.y); o[2] = f2bf_rne(a.z); o[3] = f2bf_rne(a.w);
    o[4] = f2bf_rne(b.x); o[5] = f2bf_rne(b.y); o[6] = f2bf_rne(b.z); o[7] = f2bf_rne(b.w);
    *((ushort8*)dst + i) = o;
}

// one wave per row of 256 logits: argmax(logits + g), first-index tie rule
__global__ __launch_bounds__(256)
void argmax_rows(const float* __restrict__ logits, const float* __restrict__ g,
                 int* __restrict__ idx)
{
    int wave = threadIdx.x >> 5;
    int lane = threadIdx.x & 31;
    int row  = blockIdx.x * 8 + wave;
    const float* lr = logits + (size_t)row * 256;
    const float* gr = g      + (size_t)row * 256;
    float best = -INFINITY; int bi = 0;
    #pragma unroll
    for (int s = 0; s < 8; ++s) {
        int c = lane + s * 32;
        float v = lr[c] + gr[c];
        if (v > best) { best = v; bi = c; }
    }
    #pragma unroll
    for (int off = 16; off > 0; off >>= 1) {
        float ov = __shfl_xor(best, off, 32);
        int   oi = __shfl_xor(bi,   off, 32);
        if (ov > best || (ov == best && oi < bi)) { best = ov; bi = oi; }
    }
    if (lane == 0) idx[row] = bi;
}

__global__ __launch_bounds__(256)
void gather_rows_f32(const float* __restrict__ tbl, const int* __restrict__ idx,
                     float* __restrict__ out, int outStride, int rowLen)
{
    int row = blockIdx.x;
    int r = idx[row];
    const float4* src = (const float4*)(tbl + (size_t)r * rowLen);
    float4*       dst = (float4*)(out + (size_t)row * outStride);
    for (int p = threadIdx.x; p < (rowLen >> 2); p += 256) dst[p] = src[p];
}

__global__ __launch_bounds__(256)
void gather_rows_bf16(const unsigned short* __restrict__ tbl, const int* __restrict__ idx,
                      unsigned short* __restrict__ out, int outStride, int rowLen)
{
    int row = blockIdx.x;
    int r = idx[row];
    const ushort8* src = (const ushort8*)(tbl + (size_t)r * rowLen);
    ushort8*       dst = (ushort8*)(out + (size_t)row * outStride);
    for (int p = threadIdx.x; p < (rowLen >> 3); p += 256) dst[p] = src[p];
}

__global__ __launch_bounds__(256)
void pred_error_rows(const float* __restrict__ c0, const float* __restrict__ nc0,
                     const float* __restrict__ nc1, float* __restrict__ pe)
{
    __shared__ float red[256];
    int b = blockIdx.x;
    float s = 0.f;
    for (int d = threadIdx.x; d < 4096; d += 256) {
        float cn = (d < 2048) ? nc0[(size_t)b * 2048 + d] : nc1[(size_t)b * 2048 + d - 2048];
        float t = cn - c0[(size_t)b * 4096 + d];
        s += t * t;
    }
    red[threadIdx.x] = s; __syncthreads();
    for (int off = 128; off > 0; off >>= 1) {
        if (threadIdx.x < off) red[threadIdx.x] += red[threadIdx.x + off];
        __syncthreads();
    }
    if (threadIdx.x == 0) pe[b] = red[0] * (1.0f / 4096.0f);
}

__global__ __launch_bounds__(256)
void reduce_scaled_sum(const float* __restrict__ in, float* __restrict__ out,
                       int n, float scaleF)
{
    __shared__ float red[256];
    float s = 0.f;
    for (int i = threadIdx.x; i < n; i += 256) s += in[i];
    red[threadIdx.x] = s; __syncthreads();
    for (int off = 128; off > 0; off >>= 1) {
        if (threadIdx.x < off) red[threadIdx.x] += red[threadIdx.x + off];
        __syncthreads();
    }
    if (threadIdx.x == 0) out[0] = red[0] * scaleF;
}

__global__ __launch_bounds__(256)
void confidence_kernel(const float* __restrict__ pe, const float* __restrict__ meanPe,
                       float* __restrict__ conf, float* __restrict__ cq)
{
    int b = blockIdx.x * 256 + threadIdx.x;
    float p  = pe[b];
    float pc = fabsf(p - meanPe[0]);
    float temp = 1.0f / (1.0f + expf(-pc));
    float c = 0.5f * (tanhf(-p * temp) + 1.0f);
    conf[b] = c;
    cq[b]   = c * (1.0f - c);
}

__global__ __launch_bounds__(256)
void peout_kernel(const float* __restrict__ pe, const float* __restrict__ rc,
                  float* __restrict__ out)
{
    int b = blockIdx.x * 256 + threadIdx.x;
    out[b] = pe[b] + rc[0];
}

__global__ __launch_bounds__(256)
void continue_up_kernel(const float* __restrict__ hidden, const float* __restrict__ conf,
                        float* __restrict__ out)
{
    size_t i = (size_t)blockIdx.x * 256 + threadIdx.x;
    int row = (int)(i >> 12);              // 4096 cols
    out[i] = hidden[i] * (1.0f - conf[row]);
}

extern "C" void kernel_launch(void* const* d_in, const int* in_sizes, int n_in,
                              void* d_out, int out_size, void* d_ws, size_t ws_size,
                              hipStream_t stream)
{
    (void)in_sizes; (void)n_in; (void)out_size; (void)ws_size;
    const float* x        = (const float*)d_in[0];
    const float* g_main0  = (const float*)d_in[1];
    const float* g_main1  = (const float*)d_in[2];
    const float* g_next0  = (const float*)d_in[3];
    const float* g_next1  = (const float*)d_in[4];
    const float* W_proc   = (const float*)d_in[5];
    const float* b_proc   = (const float*)d_in[6];
    const float* patterns0= (const float*)d_in[7];
    const float* patterns1= (const float*)d_in[8];
    const float* attn0_W  = (const float*)d_in[9];
    const float* attn0_b  = (const float*)d_in[10];
    const float* attn1_W  = (const float*)d_in[11];
    const float* attn1_b  = (const float*)d_in[12];
    const float* proj0_W  = (const float*)d_in[13];
    const float* proj0_b  = (const float*)d_in[14];
    const float* pen_W    = (const float*)d_in[15];
    const float* pen_b    = (const float*)d_in[16];
    const float* nW_proc  = (const float*)d_in[17];
    const float* nb_proc  = (const float*)d_in[18];
    const float* npat0    = (const float*)d_in[19];
    const float* npat1    = (const float*)d_in[20];
    const float* nattn0_W = (const float*)d_in[21];
    const float* nattn0_b = (const float*)d_in[22];
    const float* nattn1_W = (const float*)d_in[23];
    const float* nattn1_b = (const float*)d_in[24];

    float* out     = (float*)d_out;
    float* cu_out  = out;                              // 4096*4096
    float* pen_out = out + (size_t)4096 * 4096;        // 4096*2048
    float* pe_out  = pen_out + (size_t)4096 * 2048;    // 4096

    // ---- workspace layout ----
    char* wsb = (char*)d_ws;
    size_t off = 0;
    #define ALLOC_F(name, n) float* name = (float*)(wsb + off); off += (size_t)(n) * 4;
    #define ALLOC_H(name, n) unsigned short* name = (unsigned short*)(wsb + off); off += (((size_t)(n) * 2 + 15) & ~(size_t)15);
    #define ALLOC_I(name, n) int* name = (int*)(wsb + off); off += (size_t)(n) * 4;

    ALLOC_F(hidden, 16777216)          // f32 hidden (for continue_up)
    ALLOC_F(c0f,    16777216)          // f32 c0 (for pred_error)
    ALLOC_F(nc0f,   8388608)
    ALLOC_F(nc1f,   8388608)
    ALLOC_F(logits, 1048576)           // reused 4x
    ALLOC_F(pe,   4096) ALLOC_F(conf, 4096) ALLOC_F(cq, 4096) ALLOC_F(scal, 16)
    ALLOC_I(idx0, 4096) ALLOC_I(idx1, 4096) ALLOC_I(idx2, 4096) ALLOC_I(idx3, 4096)
    // bf16 operand pool
    ALLOC_H(xbf,      8388608)
    ALLOC_H(Wprocbf,  8388608)
    ALLOC_H(hidbf,    16777216)
    ALLOC_H(nWprocbf, 16777216)
    ALLOC_H(aHbf,     16777216)
    ALLOC_H(attn0Wbf, 1048576)
    ALLOC_H(attn1Wbf, 524288)
    ALLOC_H(proj0Wbf, 8388608)
    ALLOC_H(penWbf,   12582912)
    ALLOC_H(nattn0Wbf,1048576)
    ALLOC_H(nattn1Wbf,524288)
    ALLOC_H(p0bf,     1048576)
    ALLOC_H(p1bf,     524288)
    ALLOC_H(np0bf,    524288)
    ALLOC_H(np1bf,    524288)
    ALLOC_H(mcbf,     25165824)        // c0|c1 concat, stride 6144
    ALLOC_H(curbf,    8388608)
    ALLOC_H(nc0bf,    8388608)

    dim3 blk(256);
    #define CVT(src, dst, n) f32_to_bf16<<<(n)/2048, blk, 0, stream>>>(src, dst, (n)/8);
    CVT(x,        xbf,       8388608)
    CVT(W_proc,   Wprocbf,   8388608)
    CVT(nW_proc,  nWprocbf,  16777216)
    CVT(attn0_W,  attn0Wbf,  1048576)
    CVT(attn1_W,  attn1Wbf,  524288)
    CVT(proj0_W,  proj0Wbf,  8388608)
    CVT(pen_W,    penWbf,    12582912)
    CVT(nattn0_W, nattn0Wbf, 1048576)
    CVT(nattn1_W, nattn1Wbf, 524288)
    CVT(patterns0,p0bf,      1048576)
    CVT(patterns1,p1bf,      524288)
    CVT(npat0,    np0bf,     524288)
    CVT(npat1,    np1bf,     524288)

    // hidden = relu(x @ W_proc + b)            [4096,4096,2048] -> f32 + bf16
    gemm_bf16_wmma<1|4|8><<<dim3(32, 32), blk, 0, stream>>>(xbf, 2048, Wprocbf, 4096, b_proc, hidden, hidbf, 4096, 2048, nullptr);
    // a = relu(hidden @ nW_proc + nb)          [4096,4096,4096] -> bf16 only
    gemm_bf16_wmma<1|4><<<dim3(32, 32), blk, 0, stream>>>(hidbf, 4096, nWprocbf, 4096, nb_proc, nullptr, aHbf, 4096, 4096, nullptr);
    // logits0 = hidden @ attn0_W + b           [4096,256,4096]
    gemm_bf16_wmma<8><<<dim3(2, 32), blk, 0, stream>>>(hidbf, 4096, attn0Wbf, 256, attn0_b, logits, nullptr, 256, 4096, nullptr);
    argmax_rows<<<512, blk, 0, stream>>>(logits, g_main0, idx0);
    gather_rows_bf16<<<4096, blk, 0, stream>>>(p0bf, idx0, mcbf, 6144, 4096);       // c0 (bf16)
    gather_rows_f32 <<<4096, blk, 0, stream>>>(patterns0, idx0, c0f, 4096, 4096);   // c0 (f32)
    // cur = c0 @ proj0_W + b                   [4096,2048,4096] -> bf16 only
    gemm_bf16_wmma<4><<<dim3(16, 32), blk, 0, stream>>>(mcbf, 6144, proj0Wbf, 2048, proj0_b, nullptr, curbf, 2048, 4096, nullptr);
    // logits1 = cur @ attn1_W + b              [4096,256,2048]
    gemm_bf16_wmma<8><<<dim3(2, 32), blk, 0, stream>>>(curbf, 2048, attn1Wbf, 256, attn1_b, logits, nullptr, 256, 2048, nullptr);
    argmax_rows<<<512, blk, 0, stream>>>(logits, g_main1, idx1);
    gather_rows_bf16<<<4096, blk, 0, stream>>>(p1bf, idx1, mcbf + 4096, 6144, 2048); // c1 (bf16)
    // nlogits0 = a @ nattn0_W + b              [4096,256,4096]
    gemm_bf16_wmma<8><<<dim3(2, 32), blk, 0, stream>>>(aHbf, 4096, nattn0Wbf, 256, nattn0_b, logits, nullptr, 256, 4096, nullptr);
    argmax_rows<<<512, blk, 0, stream>>>(logits, g_next0, idx2);
    gather_rows_bf16<<<4096, blk, 0, stream>>>(np0bf, idx2, nc0bf, 2048, 2048);
    gather_rows_f32 <<<4096, blk, 0, stream>>>(npat0, idx2, nc0f, 2048, 2048);
    // nlogits1 = nc0 @ nattn1_W + b            [4096,256,2048]
    gemm_bf16_wmma<8><<<dim3(2, 32), blk, 0, stream>>>(nc0bf, 2048, nattn1Wbf, 256, nattn1_b, logits, nullptr, 256, 2048, nullptr);
    argmax_rows<<<512, blk, 0, stream>>>(logits, g_next1, idx3);
    gather_rows_f32<<<4096, blk, 0, stream>>>(npat1, idx3, nc1f, 2048, 2048);
    // scalar chain
    pred_error_rows<<<4096, blk, 0, stream>>>(c0f, nc0f, nc1f, pe);
    reduce_scaled_sum<<<1, blk, 0, stream>>>(pe, scal, 4096, 1.0f / 4096.0f);
    confidence_kernel<<<16, blk, 0, stream>>>(pe, scal, conf, cq);
    reduce_scaled_sum<<<1, blk, 0, stream>>>(cq, scal + 1, 4096, 0.1f / 4096.0f);
    peout_kernel<<<16, blk, 0, stream>>>(pe, scal + 1, pe_out);
    // continue_up = hidden * (1 - confidence)
    continue_up_kernel<<<65536, blk, 0, stream>>>(hidden, conf, cu_out);
    // penultimate = (mc @ pen_W + pen_b) * confidence   [4096,2048,6144]
    gemm_bf16_wmma<2|8><<<dim3(16, 32), blk, 0, stream>>>(mcbf, 6144, penWbf, 2048, pen_b, pen_out, nullptr, 2048, 6144, conf);
}
